// Fastformer_43559558316497
// MI455X (gfx1250) — matrix operand order repaired and checked by reading, compiled, not run on hardware
//
#include <hip/hip_runtime.h>
#include <hip/hip_bf16.h>

typedef __attribute__((ext_vector_type(16))) _Float16 v16h;
typedef __attribute__((ext_vector_type(8)))  _Float16 v8h;
typedef __attribute__((ext_vector_type(8)))  float    v8f;

#define B_   8
#define S_   4096
#define F_   768
#define H_   12
#define D_   64
#define NTOK (B_*S_)          // 32768
#define INV_SCALE 0.125f      // 1/sqrt(64)

// LDS byte offset of a __shared__ object: generic LDS-aperture pointers carry
// the wave-relative LDS offset in addr[31:0] (ISA 10.2 aperture mapping).
#define LDS_OFF(p) ((unsigned)(unsigned long long)(p))

// ---------------------------------------------------------------------------
// x (f32) -> x16 (f16), vectorized 8 elements/thread. 50MB result is L2-resident
// so both GEMMs read A from L2 instead of re-reading 100MB f32 from HBM.
// ---------------------------------------------------------------------------
__global__ __launch_bounds__(256) void cvt_x_f16(const float* __restrict__ X,
                                                 _Float16* __restrict__ X16) {
  size_t i = ((size_t)blockIdx.x * 256 + threadIdx.x) * 8;
  float4 f0 = *(const float4*)(X + i);
  float4 f1 = *(const float4*)(X + i + 4);
  v8h h;
  h[0] = (_Float16)f0.x; h[1] = (_Float16)f0.y;
  h[2] = (_Float16)f0.z; h[3] = (_Float16)f0.w;
  h[4] = (_Float16)f1.x; h[5] = (_Float16)f1.y;
  h[6] = (_Float16)f1.z; h[7] = (_Float16)f1.w;
  *(v8h*)(X16 + i) = h;
}

// ---------------------------------------------------------------------------
// Transpose + convert f32 W[K][N] -> f16 WT[N][K]
// ---------------------------------------------------------------------------
__global__ void cvt_transpose_f16(const float* __restrict__ W,
                                  _Float16* __restrict__ WT, int K, int N) {
  int i = blockIdx.x * blockDim.x + threadIdx.x;
  if (i >= K * N) return;
  int n = i / K;
  int k = i - n * K;
  WT[i] = (_Float16)W[(size_t)k * N + n];
}

// ---------------------------------------------------------------------------
// WMMA GEMM: Out[NTOK][768] = X16 @ WT^T   (WT is [N][K] f16)
// Block: 256 thr = 8 waves. Tile M=128 N=64 K=32, double-buffered LDS,
// tiles staged with global_load_async_to_lds_b128 (ASYNCcnt pipeline).
// ---------------------------------------------------------------------------
__global__ __launch_bounds__(256) void gemm_xw_wmma(
    const _Float16* __restrict__ X16, const _Float16* __restrict__ WT,
    float* __restrict__ Out) {
  __shared__ __align__(16) _Float16 sA[2][128 * 32];
  __shared__ __align__(16) _Float16 sB[2][64 * 32];
  const int tid  = threadIdx.x;
  const int lane = tid & 31;
  const int wv   = tid >> 5;
  const int n0   = blockIdx.x * 64;
  const int m0   = blockIdx.y * 128;

  // --- async staging addresses (per stage: 2x b128 for A, 1x b128 for B) ---
  const int aRow = tid >> 1, aSeg = (tid & 1) * 16;   // halves within row
  const int bRow = tid >> 2, bSeg = (tid & 3) * 8;
  const unsigned aG = (unsigned)(((m0 + aRow) * F_ + aSeg) * 2);  // byte voffset
  const unsigned bG = (unsigned)(((n0 + bRow) * F_ + bSeg) * 2);
  const unsigned aL[2] = { LDS_OFF(&sA[0][aRow * 32 + aSeg]),
                           LDS_OFF(&sA[1][aRow * 32 + aSeg]) };
  const unsigned bL[2] = { LDS_OFF(&sB[0][bRow * 32 + bSeg]),
                           LDS_OFF(&sB[1][bRow * 32 + bSeg]) };
  const unsigned long long xbase = (unsigned long long)X16;
  const unsigned long long wbase = (unsigned long long)WT;

#define STAGE(buf, k0)                                                        \
  do {                                                                        \
    unsigned ag = aG + (unsigned)(k0) * 2u;                                   \
    unsigned bg = bG + (unsigned)(k0) * 2u;                                   \
    asm volatile("global_load_async_to_lds_b128 %0, %1, %2 offset:0"          \
                 :: "v"(aL[buf]), "v"(ag), "s"(xbase) : "memory");            \
    asm volatile("global_load_async_to_lds_b128 %0, %1, %2 offset:16"         \
                 :: "v"(aL[buf]), "v"(ag), "s"(xbase) : "memory");            \
    asm volatile("global_load_async_to_lds_b128 %0, %1, %2 offset:0"          \
                 :: "v"(bL[buf]), "v"(bg), "s"(wbase) : "memory");            \
  } while (0)

  // --- WMMA fragment geometry (wave32 lane split per ISA 7.12.2) ---
  const int arow = lane & 15;
  const int hi   = (lane >> 4) & 1;
  const int klo  = hi ? 16 : 0;   // A: lanes 0-15 K0-15, 16-31 K16-31
  const int kb   = hi ? 16 : 0;   // B: lanes 0-15 K0-15, 16-31 K16-31

  v8f acc0 = {}, acc1 = {}, acc2 = {}, acc3 = {};

  STAGE(0, 0);
  const int NK = F_ / 32;   // 24
  for (int kt = 0; kt < NK; ++kt) {
    const int cur = kt & 1;
    if (kt + 1 < NK) {
      STAGE(cur ^ 1, (kt + 1) * 32);
      asm volatile("s_wait_asynccnt 0x3" ::: "memory");  // drain previous stage
    } else {
      asm volatile("s_wait_asynccnt 0x0" ::: "memory");
    }
    __syncthreads();   // staged tile visible to all waves

    union { v16h v; v8h h[2]; } a;
    const _Float16* ap = &sA[cur][(wv * 16 + arow) * 32 + klo];
    a.h[0] = *(const v8h*)(ap);
    a.h[1] = *(const v8h*)(ap + 8);

    const v16h b0 = *(const v16h*)(&sB[cur][( 0 + arow) * 32 + kb]);
    const v16h b1 = *(const v16h*)(&sB[cur][(16 + arow) * 32 + kb]);
    const v16h b2 = *(const v16h*)(&sB[cur][(32 + arow) * 32 + kb]);
    const v16h b3 = *(const v16h*)(&sB[cur][(48 + arow) * 32 + kb]);

    acc0 = __builtin_amdgcn_wmma_f32_16x16x32_f16(false, a.v, false, b0, (short)0, acc0, false, false);
    acc1 = __builtin_amdgcn_wmma_f32_16x16x32_f16(false, a.v, false, b1, (short)0, acc1, false, false);
    acc2 = __builtin_amdgcn_wmma_f32_16x16x32_f16(false, a.v, false, b2, (short)0, acc2, false, false);
    acc3 = __builtin_amdgcn_wmma_f32_16x16x32_f16(false, a.v, false, b3, (short)0, acc3, false, false);

    __syncthreads();   // all waves done reading buf before it is re-staged
  }
#undef STAGE

  const int rbase = m0 + wv * 16 + (hi ? 8 : 0);
  const int cbase = n0 + arow;
#pragma unroll
  for (int i = 0; i < 8; ++i) {
    size_t base = (size_t)(rbase + i) * F_ + cbase;
    Out[base +  0] = acc0[i];
    Out[base + 16] = acc1[i];
    Out[base + 32] = acc2[i];
    Out[base + 48] = acc3[i];
  }
}

// ---------------------------------------------------------------------------
// scores[b][h][s] = sum_c Src[b,s,c] * (Ctx ? Ctx[b,c] : 1) * Wa[c][h]
// One wave per token.
// ---------------------------------------------------------------------------
__global__ __launch_bounds__(256) void scores_kernel(
    const float* __restrict__ Src, const float* __restrict__ Wa,
    const float* __restrict__ Ctx, float* __restrict__ Sc) {
  int wv   = blockIdx.x * (blockDim.x >> 5) + (threadIdx.x >> 5);
  int lane = threadIdx.x & 31;
  int b = wv >> 12;
  int s = wv & (S_ - 1);
  const float* row = Src + (size_t)wv * F_;
  const float* ctx = Ctx ? (Ctx + (size_t)b * F_) : nullptr;
  float acc[H_];
#pragma unroll
  for (int h = 0; h < H_; ++h) acc[h] = 0.f;
  for (int c = lane; c < F_; c += 32) {
    float base = row[c];
    if (ctx) base *= ctx[c];
    const float* wr = Wa + (size_t)c * H_;
#pragma unroll
    for (int h = 0; h < H_; ++h) acc[h] += base * wr[h];
  }
#pragma unroll
  for (int h = 0; h < H_; ++h) {
    float v = acc[h];
    for (int off = 16; off > 0; off >>= 1) v += __shfl_xor(v, off);
    if (lane == 0) Sc[((size_t)b * H_ + h) * S_ + s] = v;
  }
}

// ---------------------------------------------------------------------------
// Pool weights + context: block per (h=blockIdx.x, b=blockIdx.y)
// ---------------------------------------------------------------------------
__global__ __launch_bounds__(256) void pool_ctx_kernel(
    const float* __restrict__ Sc, const float* __restrict__ Mask,
    const float* __restrict__ Src, const float* __restrict__ ScaleCtx,
    float* __restrict__ CtxOut) {
  __shared__ float sE[S_];
  __shared__ float sR[256];
  __shared__ float sInv;
  int h = blockIdx.x, b = blockIdx.y;
  int tid = threadIdx.x;
  const float* sc = Sc + ((size_t)b * H_ + h) * S_;
  const float* mk = Mask + (size_t)b * S_;

  float local = 0.f;
  for (int s = tid; s < S_; s += 256) {
    float e = __expf(sc[s] * INV_SCALE) * mk[s];
    sE[s] = e;
    local += e;
  }
  sR[tid] = local;
  __syncthreads();
  for (int st = 128; st > 0; st >>= 1) {
    if (tid < st) sR[tid] += sR[tid + st];
    __syncthreads();
  }
  if (tid == 0) sInv = 1.0f / (sR[0] + 1e-8f);
  __syncthreads();

  int d = tid & 63, g = tid >> 6;
  float acc = 0.f;
  const float* src = Src + (size_t)b * S_ * F_ + h * D_ + d;
  for (int s = g; s < S_; s += 4) acc += src[(size_t)s * F_] * sE[s];
  __syncthreads();
  sR[tid] = acc;
  __syncthreads();
  if (g == 0) {
    float t = sR[d] + sR[d + 64] + sR[d + 128] + sR[d + 192];
    t *= sInv;
    if (ScaleCtx) t *= ScaleCtx[((size_t)b * H_ + h) * D_ + d];
    CtxOut[((size_t)b * H_ + h) * D_ + d] = t;
  }
}

// ---------------------------------------------------------------------------
// Final: Out += (query_h * k_ctx) @ Wo  per head (Out already holds query).
// ---------------------------------------------------------------------------
__global__ __launch_bounds__(256) void final_wmma(
    float* __restrict__ Out, const float* __restrict__ Kctx,
    const _Float16* __restrict__ WoT) {
  __shared__ __align__(16) _Float16 sQ[128 * D_];
  __shared__ float sC[D_];
  int tg = blockIdx.x, h = blockIdx.y, b = blockIdx.z;
  int tid = threadIdx.x, lane = tid & 31, wv = tid >> 5;
  int tok0 = tg * 128;

  if (tid < D_) sC[tid] = Kctx[((size_t)b * H_ + h) * D_ + tid];
  __syncthreads();

  {
    int row = tid >> 1, seg = (tid & 1) * 32;
    const float* src = Out + ((size_t)(b * S_ + tok0 + row)) * F_ + h * D_ + seg;
#pragma unroll
    for (int j = 0; j < 32; ++j)
      sQ[row * D_ + seg + j] = (_Float16)(src[j] * sC[seg + j]);
  }
  __syncthreads();

  const int arow = lane & 15;
  const int hi   = (lane >> 4) & 1;
  const int klo  = hi ? 16 : 0;
  const int kb   = hi ? 16 : 0;

  v8f acc0 = {}, acc1 = {}, acc2 = {}, acc3 = {};
#pragma unroll
  for (int ks = 0; ks < 2; ++ks) {
    union { v16h v; v8h hh[2]; } a;
    const _Float16* ap = sQ + (wv * 16 + arow) * D_ + ks * 32 + klo;
    a.hh[0] = *(const v8h*)(ap);
    a.hh[1] = *(const v8h*)(ap + 8);

    const v16h b0 = *(const v16h*)(WoT + ( 0 + arow) * D_ + ks * 32 + kb);
    const v16h b1 = *(const v16h*)(WoT + (16 + arow) * D_ + ks * 32 + kb);
    const v16h b2 = *(const v16h*)(WoT + (32 + arow) * D_ + ks * 32 + kb);
    const v16h b3 = *(const v16h*)(WoT + (48 + arow) * D_ + ks * 32 + kb);

    acc0 = __builtin_amdgcn_wmma_f32_16x16x32_f16(false, a.v, false, b0, (short)0, acc0, false, false);
    acc1 = __builtin_amdgcn_wmma_f32_16x16x32_f16(false, a.v, false, b1, (short)0, acc1, false, false);
    acc2 = __builtin_amdgcn_wmma_f32_16x16x32_f16(false, a.v, false, b2, (short)0, acc2, false, false);
    acc3 = __builtin_amdgcn_wmma_f32_16x16x32_f16(false, a.v, false, b3, (short)0, acc3, false, false);
  }

  int rbase = tok0 + wv * 16 + (hi ? 8 : 0);
  int cbase = h * D_ + arow;
#pragma unroll
  for (int i = 0; i < 8; ++i) {
    size_t base = ((size_t)(b * S_ + rbase + i)) * F_ + cbase;
    Out[base +  0] += acc0[i];
    Out[base + 16] += acc1[i];
    Out[base + 32] += acc2[i];
    Out[base + 48] += acc3[i];
  }
}

// ---------------------------------------------------------------------------
extern "C" void kernel_launch(void* const* d_in, const int* in_sizes, int n_in,
                              void* d_out, int out_size, void* d_ws, size_t ws_size,
                              hipStream_t stream) {
  const float* x    = (const float*)d_in[0];
  const float* mask = (const float*)d_in[1];
  const float* Wq   = (const float*)d_in[2];
  const float* Wk   = (const float*)d_in[3];
  const float* Wqa  = (const float*)d_in[4];
  const float* Wka  = (const float*)d_in[5];
  const float* Wo   = (const float*)d_in[6];
  float* out = (float*)d_out;   // holds query, then query + kv

  char* ws = (char*)d_ws;
  size_t off = 0;
  auto alloc = [&](size_t bytes) -> void* {
    void* p = ws + off;
    off += (bytes + 255) & ~(size_t)255;
    return p;
  };
  _Float16* x16  = (_Float16*)alloc((size_t)NTOK * F_ * sizeof(_Float16));
  float*    key  = (float*)   alloc((size_t)NTOK * F_ * sizeof(float));
  _Float16* WqT  = (_Float16*)alloc((size_t)F_ * F_ * sizeof(_Float16));
  _Float16* WkT  = (_Float16*)alloc((size_t)F_ * F_ * sizeof(_Float16));
  _Float16* WoT  = (_Float16*)alloc((size_t)D_ * D_ * sizeof(_Float16));
  float*    qsc  = (float*)   alloc((size_t)B_ * H_ * S_ * sizeof(float));
  float*    ksc  = (float*)   alloc((size_t)B_ * H_ * S_ * sizeof(float));
  float*    qctx = (float*)   alloc((size_t)B_ * H_ * D_ * sizeof(float));
  float*    kctx = (float*)   alloc((size_t)B_ * H_ * D_ * sizeof(float));

  // input/weight prep (f16; weights N-major for WMMA B fragments)
  cvt_x_f16<<<(NTOK * F_) / (256 * 8), 256, 0, stream>>>(x, x16);
  cvt_transpose_f16<<<(F_ * F_ + 255) / 256, 256, 0, stream>>>(Wq, WqT, F_, F_);
  cvt_transpose_f16<<<(F_ * F_ + 255) / 256, 256, 0, stream>>>(Wk, WkT, F_, F_);
  cvt_transpose_f16<<<(D_ * D_ + 255) / 256, 256, 0, stream>>>(Wo, WoT, D_, D_);

  // query = x@Wq (into d_out), key = x@Wk   (async-to-LDS double-buffered WMMA)
  dim3 gg(F_ / 64, NTOK / 128);
  gemm_xw_wmma<<<gg, 256, 0, stream>>>(x16, WqT, out);
  gemm_xw_wmma<<<gg, 256, 0, stream>>>(x16, WkT, key);

  // q pooling
  scores_kernel<<<NTOK / 8, 256, 0, stream>>>(out, Wqa, nullptr, qsc);
  dim3 gp(H_, B_);
  pool_ctx_kernel<<<gp, 256, 0, stream>>>(qsc, mask, out, nullptr, qctx);

  // k pooling (qk never materialized)
  scores_kernel<<<NTOK / 8, 256, 0, stream>>>(key, Wka, qctx, ksc);
  pool_ctx_kernel<<<gp, 256, 0, stream>>>(ksc, mask, key, qctx, kctx);

  // out = query + (qh * k_ctx) @ Wo
  dim3 gf(S_ / 128, H_, B_);
  final_wmma<<<gf, 256, 0, stream>>>(out, kctx, WoT);
}
